// PhysiologicalGNN_62740882260650
// MI455X (gfx1250) — matrix operand order repaired and checked by reading, not compiled, not run on hardware
//
#include <hip/hip_runtime.h>
#include <hip/hip_bf16.h>

#define N_NODES 300000
#define N_EDGES 600000
#define HID 64
#define MT 4   // row tiles (16 rows each) per wave in the GEMM

typedef __attribute__((ext_vector_type(16))) __bf16 v16bf;
typedef __attribute__((ext_vector_type(8)))  __bf16 v8bf;
typedef __attribute__((ext_vector_type(8)))  float  v8f;

// Monotonic float<->uint mapping so segment-max can use integer atomicMax.
__device__ __forceinline__ unsigned enc_f32(float f) {
  unsigned b = __float_as_uint(f);
  return (b & 0x80000000u) ? ~b : (b | 0x80000000u);
}
__device__ __forceinline__ float dec_f32(unsigned u) {
  unsigned b = (u & 0x80000000u) ? (u ^ 0x80000000u) : ~u;
  return __uint_as_float(b);
}

// f32 -> bf16 streaming convert (used once for the input x).
__global__ void k_cvt(const float* __restrict__ in, __bf16* __restrict__ out, int n) {
  int t = blockIdx.x * blockDim.x + threadIdx.x;
  if (t < n) out[t] = (__bf16)in[t];
}

// Wt[n, k] = bf16(W[k, n]): transpose + convert so GEMM B-fragments are
// 16 consecutive bf16 (one 32-byte load per lane per k-step).
__global__ void k_cvt_wt(const float* __restrict__ W, __bf16* __restrict__ Wt,
                         int K, int ncol) {
  int t = blockIdx.x * blockDim.x + threadIdx.x;
  if (t >= K * ncol) return;
  int n = t / K, k = t % K;
  Wt[t] = (__bf16)W[(size_t)k * ncol + n];
}

// Wae[d*H+h] = sum_c We[d, h*C+c] * ae[h,c]   (collapses edge-feature GEMM)
__global__ void k_wae(const float* __restrict__ We, const float* __restrict__ ae,
                      float* __restrict__ Wae, int H, int HC) {
  int t = threadIdx.x;
  if (t >= 3 * H) return;
  int d = t / H, h = t % H;
  float s = 0.f;
  for (int c = 0; c < HID; ++c) s += We[d * HC + h * HID + c] * ae[h * HID + c];
  Wae[d * H + h] = s;
}

// Hout[N, ncol] = A[N, K](bf16) @ Wt[ncol, K](bf16)^T, f32 accumulate.
// Block = 128 threads = 4 waves; wave w owns cols [by*64 + w*16, +16),
// MT=4 row tiles of 16 -> 64 rows per block, B fragment reused 4x.
__global__ __launch_bounds__(128)
void k_gemm(const __bf16* __restrict__ A, const __bf16* __restrict__ Wt,
            float* __restrict__ Hout, int K, int ncol) {
  const int lane = threadIdx.x & 31;
  const int wave = threadIdx.x >> 5;
  const int m0   = blockIdx.x * (16 * MT);
  const int n0   = blockIdx.y * 64 + wave * 16;
  const int half = lane >> 4;   // 0: lanes 0-15, 1: lanes 16-31
  const int l15  = lane & 15;
  const int colB = n0 + l15;

  v8f acc[MT] = {};
  // B: lane holds col colB, K = k0 + half*16 + {0..15} -> contiguous in Wt.
  const __bf16* bptr = Wt + (size_t)colB * K + half * 16;
  // A: lane holds row (clamped for the ragged last block; stores are guarded).
  const __bf16* aptr[MT];
#pragma unroll
  for (int rt = 0; rt < MT; ++rt) {
    int r = m0 + rt * 16 + l15;
    if (r >= N_NODES) r = N_NODES - 1;
    aptr[rt] = A + (size_t)r * K + half * 16;
  }

  for (int k0 = 0; k0 < K; k0 += 32) {
    v16bf b = *(const v16bf*)(bptr + k0);
#pragma unroll
    for (int rt = 0; rt < MT; ++rt) {
      // A fragment: elems 0..15 -> K = k0+half*16+{0..15}.
      v8bf lo = *(const v8bf*)(aptr[rt] + k0);
      v8bf hi = *(const v8bf*)(aptr[rt] + k0 + 8);
      v16bf a = __builtin_shufflevector(lo, hi, 0, 1, 2, 3, 4, 5, 6, 7,
                                        8, 9, 10, 11, 12, 13, 14, 15);
      acc[rt] = __builtin_amdgcn_wmma_f32_16x16x32_bf16(
          false, a, false, b, (short)0, acc[rt], false, false);
    }
  }

#pragma unroll
  for (int rt = 0; rt < MT; ++rt) {
    int mb = m0 + rt * 16;
    if (mb < N_NODES) {   // N % 16 == 0: tiles fully in or fully out
      float* orow = Hout + (size_t)(mb + half * 8) * ncol + colB;
#pragma unroll
      for (int j = 0; j < 8; ++j) orow[(size_t)j * ncol] = acc[rt][j];
    }
  }
}

// Per (node, head) attention logits: asrc = <h, a_s>, adst = <h, a_d>.
__global__ void k_alpha(const float* __restrict__ Hb, const float* __restrict__ as_,
                        const float* __restrict__ ad_, float* __restrict__ asrc,
                        float* __restrict__ adst, int H) {
  int t = blockIdx.x * blockDim.x + threadIdx.x;
  if (t >= N_NODES * H) return;
  int n = t / H, h = t % H;
  const float4* hv = (const float4*)(Hb + (size_t)n * H * HID + h * HID);
  const float4* sv = (const float4*)(as_ + h * HID);
  const float4* dv = (const float4*)(ad_ + h * HID);
  float ss = 0.f, sd = 0.f;
#pragma unroll
  for (int i = 0; i < HID / 4; ++i) {
    float4 hh = hv[i], aa = sv[i], bb = dv[i];
    ss += hh.x * aa.x + hh.y * aa.y + hh.z * aa.z + hh.w * aa.w;
    sd += hh.x * bb.x + hh.y * bb.y + hh.z * bb.z + hh.w * bb.w;
  }
  asrc[t] = ss;
  adst[t] = sd;
}

__global__ void k_init(float* __restrict__ acc, unsigned* __restrict__ amax,
                       float* __restrict__ denom, int nhc, int nh) {
  int t = blockIdx.x * blockDim.x + threadIdx.x;
  if (t < nhc) acc[t] = 0.f;
  if (t < nh) { amax[t] = 0u; denom[t] = 0.f; }
}

// Per-edge logits + leaky-relu + segment max (uint-encoded atomicMax).
__global__ void k_edge_alpha(const int* __restrict__ src, const int* __restrict__ dst,
                             const float* __restrict__ ea, const float* __restrict__ asrc,
                             const float* __restrict__ adst, const float* __restrict__ Wae,
                             float* __restrict__ alphaE, unsigned* __restrict__ amax, int H) {
  int e = blockIdx.x * blockDim.x + threadIdx.x;
  if (e >= N_EDGES) return;
  int s = src[e], d = dst[e];
  float e0 = ea[e * 3 + 0], e1 = ea[e * 3 + 1], e2 = ea[e * 3 + 2];
  for (int h = 0; h < H; ++h) {
    float al = asrc[s * H + h] + adst[d * H + h]
             + e0 * Wae[h] + e1 * Wae[H + h] + e2 * Wae[2 * H + h];
    al = (al > 0.f) ? al : 0.2f * al;   // leaky_relu(0.2)
    alphaE[e * H + h] = al;
    atomicMax(&amax[d * H + h], enc_f32(al));
  }
}

// Per (edge, head): ex = exp(al - max); denom[dst] += ex; acc[dst] += ex * h[src].
__global__ __launch_bounds__(128)
void k_edge_scatter(const int* __restrict__ src, const int* __restrict__ dst,
                    const float* __restrict__ alphaE, const unsigned* __restrict__ amax,
                    float* __restrict__ denom, const float* __restrict__ Hb,
                    float* __restrict__ acc, int H) {
  int t = blockIdx.x * blockDim.x + threadIdx.x;
  if (t >= N_EDGES * H) return;
  int e = t / H, h = t % H;
  int s = src[e], d = dst[e];
  float ex = __expf(alphaE[t] - dec_f32(amax[d * H + h]));
  atomicAdd(&denom[d * H + h], ex);
  const float4* hs = (const float4*)(Hb + (size_t)s * H * HID + h * HID);
  float* od = acc + (size_t)d * H * HID + h * HID;
#pragma unroll
  for (int i = 0; i < HID / 4; ++i) {
    float4 v = hs[i];
    atomicAdd(&od[i * 4 + 0], v.x * ex);
    atomicAdd(&od[i * 4 + 1], v.y * ex);
    atomicAdd(&od[i * 4 + 2], v.z * ex);
    atomicAdd(&od[i * 4 + 3], v.w * ex);
  }
}

// Normalize by denom, add bias. Intermediate layers: ELU + write bf16 (next
// GEMM input). Last layer (H=1, mean == identity): write f32 to d_out.
__global__ void k_finalize(const float* __restrict__ acc, const float* __restrict__ denom,
                           const float* __restrict__ bias, float* __restrict__ outf,
                           __bf16* __restrict__ outb, int H, int HC, int last) {
  int t = blockIdx.x * blockDim.x + threadIdx.x;
  if (t >= N_NODES * HC) return;
  int n = t / HC, r = t % HC, h = r / HID;
  float v = acc[t] / (denom[n * H + h] + 1e-16f) + bias[r];
  if (last) {
    outf[t] = v;
  } else {
    v = (v > 0.f) ? v : (__expf(v) - 1.f);   // ELU
    outb[t] = (__bf16)v;
  }
}

extern "C" void kernel_launch(void* const* d_in, const int* in_sizes, int n_in,
                              void* d_out, int out_size, void* d_ws, size_t ws_size,
                              hipStream_t stream) {
  (void)in_sizes; (void)n_in; (void)out_size; (void)ws_size;
  const float* x   = (const float*)d_in[0];
  const int*   src = (const int*)d_in[1];
  const int*   dst = src + N_EDGES;
  const float* ea  = (const float*)d_in[2];

  char* w = (char*)d_ws;
  float*    Xf     = (float*)w;    w += (size_t)N_NODES * 256 * 4;  // accumulator
  float*    Hb     = (float*)w;    w += (size_t)N_NODES * 256 * 4;  // GEMM out h (f32)
  __bf16*   Xb     = (__bf16*)w;   w += (size_t)N_NODES * 256 * 2;  // activations (bf16)
  __bf16*   Wt     = (__bf16*)w;   w += (size_t)256 * 256 * 2;      // W^T (bf16)
  float*    asrc   = (float*)w;    w += (size_t)N_NODES * 4 * 4;
  float*    adst   = (float*)w;    w += (size_t)N_NODES * 4 * 4;
  float*    denom  = (float*)w;    w += (size_t)N_NODES * 4 * 4;
  unsigned* amax   = (unsigned*)w; w += (size_t)N_NODES * 4 * 4;
  float*    alphaE = (float*)w;    w += (size_t)N_EDGES * 4 * 4;
  float*    Wae    = (float*)w;

  // Layer-0 input -> bf16 once.
  k_cvt<<<(N_NODES * HID + 255) / 256, 256, 0, stream>>>(x, Xb, N_NODES * HID);

  int K = HID;
  for (int i = 0; i < 4; ++i) {
    const float* W   = (const float*)d_in[3 + 6 * i + 0];
    const float* as_ = (const float*)d_in[3 + 6 * i + 1];
    const float* ad_ = (const float*)d_in[3 + 6 * i + 2];
    const float* We  = (const float*)d_in[3 + 6 * i + 3];
    const float* ae  = (const float*)d_in[3 + 6 * i + 4];
    const float* b   = (const float*)d_in[3 + 6 * i + 5];
    const int H  = (i < 3) ? 4 : 1;
    const int HC = H * HID;
    const int nh = N_NODES * H, nhc = N_NODES * HC;

    k_cvt_wt<<<(K * HC + 255) / 256, 256, 0, stream>>>(W, Wt, K, HC);
    k_wae<<<1, 32, 0, stream>>>(We, ae, Wae, H, HC);
    k_gemm<<<dim3((N_NODES + 16 * MT - 1) / (16 * MT), HC / 64), 128, 0, stream>>>(
        Xb, Wt, Hb, K, HC);
    k_alpha<<<(nh + 255) / 256, 256, 0, stream>>>(Hb, as_, ad_, asrc, adst, H);
    k_init<<<(nhc + 255) / 256, 256, 0, stream>>>(Xf, amax, denom, nhc, nh);
    k_edge_alpha<<<(N_EDGES + 255) / 256, 256, 0, stream>>>(src, dst, ea, asrc, adst,
                                                            Wae, alphaE, amax, H);
    k_edge_scatter<<<(N_EDGES * H + 127) / 128, 128, 0, stream>>>(src, dst, alphaE, amax,
                                                                  denom, Hb, Xf, H);
    k_finalize<<<(nhc + 255) / 256, 256, 0, stream>>>(Xf, denom, b, (float*)d_out, Xb,
                                                      H, HC, (i < 3) ? 0 : 1);
    K = HC;   // next layer consumes the bf16 activations written by finalize
  }
}
